// MultiHeadSelfAttention_65051574665350
// MI455X (gfx1250) — compile-verified, matrix operand order checked by reading, not run on hardware
//
#include <hip/hip_runtime.h>
#include <hip/hip_bf16.h>
#include <stdint.h>

// Problem constants (from reference): B=4, S=2048, E=1024, H=16, D=64
#define B_ 4
#define S_ 2048
#define E_ 1024
#define H_ 16
#define D_ 64
#define M_ (B_ * S_)   // 8192 total rows

typedef __bf16 bf16;
typedef __attribute__((ext_vector_type(4)))  __bf16 bf16x4;
typedef __attribute__((ext_vector_type(8)))  __bf16 bf16x8;
typedef __attribute__((ext_vector_type(16))) __bf16 v16bf;
typedef __attribute__((ext_vector_type(8)))  float  v8f;
typedef __attribute__((ext_vector_type(4)))  int    v4i_;

#if __has_builtin(__builtin_amdgcn_global_load_async_to_lds_b128) && \
    __has_builtin(__builtin_amdgcn_s_wait_asynccnt)
#define HAVE_ASYNC_LDS 1
#else
#define HAVE_ASYNC_LDS 0
#endif

#if HAVE_ASYNC_LDS
// CDNA5 async VMEM->LDS copy (no VGPR round trip, tracked by ASYNCcnt).
// Builtin types its pointers by transfer width: b128 => pointer-to-int4.
typedef __attribute__((address_space(1))) v4i_ as1_v4i;
typedef __attribute__((address_space(3))) v4i_ as3_v4i;
static __device__ __forceinline__ void async_copy16(const bf16* g, bf16* l) {
  __builtin_amdgcn_global_load_async_to_lds_b128(
      (as1_v4i*)g, (as3_v4i*)l, /*offset=*/0, /*cpol=*/0);
}
#endif

// Build a 16-element WMMA operand register from two 16B chunks.
// CDNA5 16-bit A-operand layout (16x32): lane l<16 holds row l, K = kb..kb+7 in
// elems 0..7 and K = kb+16..kb+23 in elems 8..15 (kb=0); lanes 16..31 hold the
// same rows with kb=8. B operand mirrors this with "row" = output column.
static __device__ __forceinline__ v16bf load_ab(const bf16* p0, const bf16* p1) {
  bf16x8 lo = *(const bf16x8*)p0;
  bf16x8 hi = *(const bf16x8*)p1;
  return __builtin_shufflevector(lo, hi, 0,1,2,3,4,5,6,7,8,9,10,11,12,13,14,15);
}

static __device__ __forceinline__ v8f wmma_bf16(v16bf a, v16bf b, v8f c) {
  return __builtin_amdgcn_wmma_f32_16x16x32_bf16(
      /*neg_a=*/false, a, /*neg_b=*/false, b,
      /*c_mod=*/(short)0, c, /*reuse_a=*/false, /*reuse_b=*/false);
}

static __device__ __forceinline__ v8f v8f_zero() {
  v8f z;
#pragma unroll
  for (int i = 0; i < 8; ++i) z[i] = 0.0f;
  return z;
}

// ---------------------------------------------------------------------------
// Kernel: fp32 -> bf16 conversion, 4 elements per thread
// ---------------------------------------------------------------------------
__global__ void cvt_f32_to_bf16(const float* __restrict__ in,
                                bf16* __restrict__ out, int n4) {
  int i = blockIdx.x * blockDim.x + threadIdx.x;
  if (i < n4) {
    float4 v = reinterpret_cast<const float4*>(in)[i];
    bf16x4 o;
    o[0] = (bf16)v.x; o[1] = (bf16)v.y; o[2] = (bf16)v.z; o[3] = (bf16)v.w;
    reinterpret_cast<bf16x4*>(out)[i] = o;
  }
}

// ---------------------------------------------------------------------------
// Kernel: tiled bf16 GEMM  C[m,n] = sum_k A[m,k] * W[n,k]  (+ bias, epilogues)
//   mode 0: store bf16 head-major  [B,H,S,D]   (Q, K)
//   mode 1: store bf16 transposed  [B,H,D,S]   (V^T)
//   mode 2: store f32  resid[m,n] + C + bias   (output projection, pre-LN)
// Block: 128x128 C tile, 256 threads = 8 waves, wave tile 32x64.
// Double-buffered LDS, staged with async VMEM->LDS copies.
// ---------------------------------------------------------------------------
__launch_bounds__(256)
__global__ void gemm_bf16(const bf16* __restrict__ A,     // [M_, E_]
                          const bf16* __restrict__ W,     // [E_, E_], row = out feature
                          const float* __restrict__ bias, // [E_]
                          const float* __restrict__ resid,// [M_, E_] (mode 2) or null
                          void* __restrict__ out, int mode) {
  __shared__ bf16 As[2][128][40];   // 32 K + 8 pad -> 80B row stride, 16B aligned
  __shared__ bf16 Bs[2][128][40];

  const int t    = threadIdx.x;
  const int lane = t & 31;
  const int w    = t >> 5;       // 0..7
  const int wm   = w & 3;        // 4 M strips of 32
  const int wn   = w >> 2;       // 2 N strips of 64
  const int lr   = lane & 15;
  const int hi   = lane >> 4;    // half-wave select
  const int kb1  = hi * 8;       // K stripe base within chunk

  const int m0 = blockIdx.y * 128;
  const int n0 = blockIdx.x * 128;

  v8f acc[2][4];
#pragma unroll
  for (int i = 0; i < 2; ++i)
#pragma unroll
    for (int j = 0; j < 4; ++j) acc[i][j] = v8f_zero();

  const int ldr = t >> 1;          // cooperative-load row 0..127
  const int lc  = (t & 1) * 16;    // col half 0 / 16

  // Stage one 128x32 A block + 128x32 B block into LDS buffer `buf`.
  auto stage = [&](int buf, int kk) {
    const bf16* ag = &A[(size_t)(m0 + ldr) * E_ + kk + lc];
    const bf16* bg = &W[(size_t)(n0 + ldr) * E_ + kk + lc];
#if HAVE_ASYNC_LDS
    async_copy16(ag,     &As[buf][ldr][lc]);
    async_copy16(ag + 8, &As[buf][ldr][lc + 8]);
    async_copy16(bg,     &Bs[buf][ldr][lc]);
    async_copy16(bg + 8, &Bs[buf][ldr][lc + 8]);
#else
    bf16x8 ta0 = *(const bf16x8*)(ag);
    bf16x8 ta1 = *(const bf16x8*)(ag + 8);
    bf16x8 tb0 = *(const bf16x8*)(bg);
    bf16x8 tb1 = *(const bf16x8*)(bg + 8);
    if (kk + 32 < E_) {
      __builtin_prefetch(ag + 32, 0, 3);
      __builtin_prefetch(bg + 32, 0, 3);
    }
    *(bf16x8*)&As[buf][ldr][lc]     = ta0;
    *(bf16x8*)&As[buf][ldr][lc + 8] = ta1;
    *(bf16x8*)&Bs[buf][ldr][lc]     = tb0;
    *(bf16x8*)&Bs[buf][ldr][lc + 8] = tb1;
#endif
  };

  stage(0, 0);

  for (int kk = 0; kk < E_; kk += 32) {
    const int cur = (kk >> 5) & 1;
    const bool has_next = (kk + 32) < E_;
    if (has_next) stage(cur ^ 1, kk + 32);
#if HAVE_ASYNC_LDS
    // Current buffer's 4 async loads (per wave) are complete when at most the
    // next buffer's 4 remain outstanding.
    if (has_next) __builtin_amdgcn_s_wait_asynccnt(4);
    else          __builtin_amdgcn_s_wait_asynccnt(0);
#endif
    __syncthreads();

    v16bf av[2], bv[4];
#pragma unroll
    for (int i = 0; i < 2; ++i) {
      const bf16* p = &As[cur][wm * 32 + 16 * i + lr][kb1];
      av[i] = load_ab(p, p + 16);
    }
#pragma unroll
    for (int j = 0; j < 4; ++j) {
      const bf16* p = &Bs[cur][wn * 64 + 16 * j + lr][kb1];
      bv[j] = load_ab(p, p + 16);
    }
#pragma unroll
    for (int i = 0; i < 2; ++i)
#pragma unroll
      for (int j = 0; j < 4; ++j) acc[i][j] = wmma_bf16(av[i], bv[j], acc[i][j]);
    __syncthreads();
  }

  // Epilogue. C/D layout: VGPR g, lane -> row = g + 8*hi, col = lr (within tile)
#pragma unroll
  for (int i = 0; i < 2; ++i) {
#pragma unroll
    for (int j = 0; j < 4; ++j) {
      const int n  = n0 + wn * 64 + 16 * j + lr;
      const float bv = bias[n];
#pragma unroll
      for (int g = 0; g < 8; ++g) {
        const int m = m0 + wm * 32 + 16 * i + g + 8 * hi;
        float v = acc[i][j][g] + bv;
        if (mode == 2) {
          reinterpret_cast<float*>(out)[(size_t)m * E_ + n] =
              resid[(size_t)m * E_ + n] + v;
        } else {
          const int b = m >> 11, s = m & (S_ - 1);
          const int h = n >> 6,  d = n & (D_ - 1);
          size_t addr;
          if (mode == 0) addr = (((size_t)(b * H_ + h)) * S_ + s) * D_ + d;
          else           addr = (((size_t)(b * H_ + h)) * D_ + d) * S_ + s;
          reinterpret_cast<bf16*>(out)[addr] = (bf16)v;
        }
      }
    }
  }
}

// ---------------------------------------------------------------------------
// Kernel: flash attention. Grid (B*H, S/128), 256 threads = 8 waves,
// each wave owns a 16x64 Q tile and streams all S keys in 64-chunks.
// K / V^T for one (b,h) are 256 KB each in bf16 and stay L2-resident.
// ---------------------------------------------------------------------------
__launch_bounds__(256)
__global__ void attn_flash(const bf16* __restrict__ Q,    // [B,H,S,D]
                           const bf16* __restrict__ K,    // [B,H,S,D]
                           const bf16* __restrict__ Vt,   // [B,H,D,S]
                           const float* __restrict__ mask,// [B,S] additive
                           bf16* __restrict__ ctx) {      // [M_, E_]
  __shared__ bf16 Plds[8][16][72];  // wave-private P tile, padded rows (144B)

  const int t    = threadIdx.x;
  const int lane = t & 31;
  const int w    = t >> 5;
  const int lr   = lane & 15;
  const int hi   = lane >> 4;
  const int kb1  = hi * 8;

  const int bh = blockIdx.x;               // 0..63
  const int b  = bh >> 4;
  const int h  = bh & 15;
  const int q0 = blockIdx.y * 128 + w * 16;

  const bf16* Qh = Q  + (size_t)bh * S_ * D_;
  const bf16* Kh = K  + (size_t)bh * S_ * D_;
  const bf16* Vh = Vt + (size_t)bh * D_ * S_;

  // Q tile as A operand, kept resident for the whole pass (2 chunks of K=32)
  v16bf qa[2];
#pragma unroll
  for (int c = 0; c < 2; ++c) {
    const bf16* p = Qh + (size_t)(q0 + lr) * D_ + 32 * c + kb1;
    qa[c] = load_ab(p, p + 16);
  }

  v8f ctxa[4];
#pragma unroll
  for (int n = 0; n < 4; ++n) ctxa[n] = v8f_zero();
  float rmax[8], rsum[8];
#pragma unroll
  for (int g = 0; g < 8; ++g) { rmax[g] = -3.0e38f; rsum[g] = 0.0f; }

  for (int kb = 0; kb < S_; kb += 64) {
    // ---- scores S = Q K^T * (1/sqrt(D)) + mask : 4 N-tiles of 16 keys ----
    v8f s[4];
#pragma unroll
    for (int n = 0; n < 4; ++n) {
      const bf16* kp = Kh + (size_t)(kb + n * 16 + lr) * D_ + kb1;
      v16bf b0 = load_ab(kp,      kp + 16);
      v16bf b1 = load_ab(kp + 32, kp + 48);
      v8f z = v8f_zero();
      z = wmma_bf16(qa[0], b0, z);
      z = wmma_bf16(qa[1], b1, z);
      s[n] = z;
    }
#pragma unroll
    for (int n = 0; n < 4; ++n) {
      const float mv = mask[(size_t)b * S_ + kb + n * 16 + lr];
#pragma unroll
      for (int g = 0; g < 8; ++g) s[n][g] = s[n][g] * 0.125f + mv;
    }

    // ---- online softmax (row = g + 8*hi, cols live across 16-lane half) ----
    float nm[8];
#pragma unroll
    for (int g = 0; g < 8; ++g)
      nm[g] = fmaxf(fmaxf(s[0][g], s[1][g]), fmaxf(s[2][g], s[3][g]));
#pragma unroll
    for (int off = 1; off < 16; off <<= 1)
#pragma unroll
      for (int g = 0; g < 8; ++g) nm[g] = fmaxf(nm[g], __shfl_xor(nm[g], off, 32));

    float corr[8];
#pragma unroll
    for (int g = 0; g < 8; ++g) {
      float mn = fmaxf(rmax[g], nm[g]);
      corr[g] = __expf(rmax[g] - mn);
      rmax[g] = mn;
    }
    float rs[8];
#pragma unroll
    for (int g = 0; g < 8; ++g) rs[g] = 0.0f;
#pragma unroll
    for (int n = 0; n < 4; ++n)
#pragma unroll
      for (int g = 0; g < 8; ++g) {
        float p = __expf(s[n][g] - rmax[g]);
        s[n][g] = p;
        rs[g] += p;
      }
#pragma unroll
    for (int off = 1; off < 16; off <<= 1)
#pragma unroll
      for (int g = 0; g < 8; ++g) rs[g] += __shfl_xor(rs[g], off, 32);
#pragma unroll
    for (int g = 0; g < 8; ++g) rsum[g] = rsum[g] * corr[g] + rs[g];
#pragma unroll
    for (int n = 0; n < 4; ++n)
#pragma unroll
      for (int g = 0; g < 8; ++g) ctxa[n][g] *= corr[g];

    // ---- stage P (D-layout) into LDS, reload in A-layout (wave-private) ----
#pragma unroll
    for (int n = 0; n < 4; ++n)
#pragma unroll
      for (int g = 0; g < 8; ++g)
        Plds[w][g + 8 * hi][n * 16 + lr] = (bf16)s[n][g];
    __builtin_amdgcn_wave_barrier();   // LDS ops are in-order within a wave
    asm volatile("" ::: "memory");

    v16bf pa[2];
#pragma unroll
    for (int c = 0; c < 2; ++c) {
      const bf16* p = &Plds[w][lr][32 * c + kb1];
      pa[c] = load_ab(p, p + 16);
    }

    // ---- ctx += P * V  (B operand rows = V^T rows, contiguous in s) ----
#pragma unroll
    for (int n = 0; n < 4; ++n) {
      const bf16* vp = Vh + (size_t)(n * 16 + lr) * S_ + kb + kb1;
      v16bf v0 = load_ab(vp,      vp + 16);
      v16bf v1 = load_ab(vp + 32, vp + 48);
      ctxa[n] = wmma_bf16(pa[0], v0, ctxa[n]);
      ctxa[n] = wmma_bf16(pa[1], v1, ctxa[n]);
    }
  }

  // ---- normalize and store ctx in [B,S,E] bf16 ----
#pragma unroll
  for (int n = 0; n < 4; ++n)
#pragma unroll
    for (int g = 0; g < 8; ++g) {
      const int row = q0 + g + 8 * hi;
      const int col = h * D_ + n * 16 + lr;
      ctx[((size_t)b * S_ + row) * E_ + col] = (bf16)(ctxa[n][g] / rsum[g]);
    }
}

// ---------------------------------------------------------------------------
// Kernel: row-wise LayerNorm over E=1024 (one 256-thread block per row)
// ---------------------------------------------------------------------------
__launch_bounds__(256)
__global__ void layernorm_row(const float* __restrict__ x,
                              const float* __restrict__ gw,
                              const float* __restrict__ gb,
                              float* __restrict__ out) {
  __shared__ float red0[8], red1[8];
  const int row = blockIdx.x;
  const float* xr = x + (size_t)row * E_;

  float s = 0.0f, s2 = 0.0f;
  for (int c = threadIdx.x; c < E_; c += 256) {
    float v = xr[c];
    s += v; s2 += v * v;
  }
#pragma unroll
  for (int off = 1; off < 32; off <<= 1) {
    s  += __shfl_xor(s,  off, 32);
    s2 += __shfl_xor(s2, off, 32);
  }
  if ((threadIdx.x & 31) == 0) {
    red0[threadIdx.x >> 5] = s;
    red1[threadIdx.x >> 5] = s2;
  }
  __syncthreads();
  float ts = 0.0f, ts2 = 0.0f;
#pragma unroll
  for (int i = 0; i < 8; ++i) { ts += red0[i]; ts2 += red1[i]; }
  const float mu  = ts * (1.0f / E_);
  const float var = ts2 * (1.0f / E_) - mu * mu;
  const float r   = rsqrtf(var + 1e-12f);
  for (int c = threadIdx.x; c < E_; c += 256)
    out[(size_t)row * E_ + c] = (xr[c] - mu) * r * gw[c] + gb[c];
}

// ---------------------------------------------------------------------------
// Host launcher
// ---------------------------------------------------------------------------
extern "C" void kernel_launch(void* const* d_in, const int* in_sizes, int n_in,
                              void* d_out, int out_size, void* d_ws, size_t ws_size,
                              hipStream_t stream) {
  const float* X     = (const float*)d_in[0];
  const float* mask  = (const float*)d_in[1];
  const float* WQw   = (const float*)d_in[2];
  const float* WQbia = (const float*)d_in[3];
  const float* WKw   = (const float*)d_in[4];
  const float* WKbia = (const float*)d_in[5];
  const float* WVw   = (const float*)d_in[6];
  const float* WVbia = (const float*)d_in[7];
  const float* WOw   = (const float*)d_in[8];
  const float* WObia = (const float*)d_in[9];
  const float* lnw   = (const float*)d_in[10];
  const float* lnb   = (const float*)d_in[11];

  uint8_t* ws = (uint8_t*)d_ws;
  // Workspace layout (bytes)
  bf16*  Xb    = (bf16*)(ws + 0);           // 16,777,216
  bf16*  WQh   = (bf16*)(ws + 16777216);    //  2,097,152
  bf16*  WKh   = (bf16*)(ws + 18874368);
  bf16*  WVh   = (bf16*)(ws + 20971520);
  bf16*  WOh   = (bf16*)(ws + 23068672);
  bf16*  Qd    = (bf16*)(ws + 25165824);    // 16,777,216  [B,H,S,D]
  bf16*  Kd    = (bf16*)(ws + 41943040);    // 16,777,216  [B,H,S,D]
  bf16*  Vtd   = (bf16*)(ws + 58720256);    // 16,777,216  [B,H,D,S]
  bf16*  Ctx   = (bf16*)(ws + 75497472);    // 16,777,216  [M,E]
  float* preLN = (float*)(ws + 92274688);   // 33,554,432  [M,E]

  // 1) fp32 -> bf16 conversions
  {
    int n4 = (M_ * E_) / 4;
    cvt_f32_to_bf16<<<(n4 + 255) / 256, 256, 0, stream>>>(X, Xb, n4);
    int w4 = (E_ * E_) / 4;
    cvt_f32_to_bf16<<<(w4 + 255) / 256, 256, 0, stream>>>(WQw, WQh, w4);
    cvt_f32_to_bf16<<<(w4 + 255) / 256, 256, 0, stream>>>(WKw, WKh, w4);
    cvt_f32_to_bf16<<<(w4 + 255) / 256, 256, 0, stream>>>(WVw, WVh, w4);
    cvt_f32_to_bf16<<<(w4 + 255) / 256, 256, 0, stream>>>(WOw, WOh, w4);
  }

  // 2) Q/K/V projections (V stored transposed per head)
  dim3 gg(E_ / 128, M_ / 128);   // (8, 64)
  gemm_bf16<<<gg, 256, 0, stream>>>(Xb, WQh, WQbia, nullptr, Qd,  0);
  gemm_bf16<<<gg, 256, 0, stream>>>(Xb, WKh, WKbia, nullptr, Kd,  0);
  gemm_bf16<<<gg, 256, 0, stream>>>(Xb, WVh, WVbia, nullptr, Vtd, 1);

  // 3) fused flash attention
  attn_flash<<<dim3(B_ * H_, S_ / 128), 256, 0, stream>>>(Qd, Kd, Vtd, mask, Ctx);

  // 4) output projection + residual (f32)
  gemm_bf16<<<gg, 256, 0, stream>>>(Ctx, WOh, WObia, X, preLN, 2);

  // 5) LayerNorm
  layernorm_row<<<M_, 256, 0, stream>>>(preLN, lnw, lnb, (float*)d_out);
}
